// LAConv2D_10015863734306
// MI455X (gfx1250) — matrix operand order repaired and checked by reading, not compiled, not run on hardware
//
#include <hip/hip_runtime.h>

typedef __attribute__((ext_vector_type(16))) _Float16 v16h;
typedef __attribute__((ext_vector_type(8)))  float    v8f;

#define Bn 8
#define Nc 32
#define Mc 32
#define HW 192
#define HW2 (HW * HW)   // 36864

static __device__ __forceinline__ int clampi(int v, int lo, int hi) {
    return v < lo ? lo : (v > hi ? hi : v);
}

// ---------------------------------------------------------------------------
// Kernel P1: pack conv weights into WMMA A-fragment layout (f16).
// A 16x32 layout (contiguous k, matching xpack B): lanes 0-15 -> K {0..15},
// lanes 16-31 -> K {16..31}; half pair per VGPR. One wave packs:
//   wpack  : main conv weight [M=32 rows -> 2 fragments] x 9 positions
//   a1pack : attention conv1 weight [9 rows, zero-padded to 16] x 9 positions
// ---------------------------------------------------------------------------
__global__ __launch_bounds__(32) void pack_kernel(const float* __restrict__ weight,
                                                  const float* __restrict__ a1w,
                                                  _Float16* __restrict__ wpack,
                                                  _Float16* __restrict__ a1pack)
{
    const int lane = threadIdx.x;
    const int row = lane & 15, hi = lane >> 4;
    for (int p = 0; p < 9; ++p) {
        for (int mh = 0; mh < 2; ++mh)
#pragma unroll
            for (int v = 0; v < 8; ++v)
#pragma unroll
                for (int sub = 0; sub < 2; ++sub) {
                    const int k = 2 * v + sub + (hi ? 16 : 0);
                    const int m = mh * 16 + row;
                    wpack[((size_t)(p * 2 + mh) * 32 + lane) * 16 + (v * 2 + sub)] =
                        (_Float16)weight[(m * Nc + k) * 9 + p];
                }
#pragma unroll
        for (int v = 0; v < 8; ++v)
#pragma unroll
            for (int sub = 0; sub < 2; ++sub) {
                const int k = 2 * v + sub + (hi ? 16 : 0);
                const float val = (row < 9) ? a1w[(row * Nc + k) * 9 + p] : 0.0f;
                a1pack[((size_t)p * 32 + lane) * 16 + (v * 2 + sub)] =
                    (_Float16)val;
            }
    }
}

// ---------------------------------------------------------------------------
// Kernel P2: repack x (f32, channel-major) into channel-interleaved f16:
// xpack[(b*2+khi)*HW2 + pix][16 channels]. One thread per (b,khi,pix);
// strided-coalesced reads, contiguous 32B writes.
// ---------------------------------------------------------------------------
__global__ __launch_bounds__(256) void xpack_kernel(const float* __restrict__ x,
                                                    _Float16* __restrict__ xpack)
{
    const size_t idx = (size_t)blockIdx.x * 256 + threadIdx.x;  // 0 .. B*2*HW2
    const int pix = (int)(idx % HW2);
    const int bk  = (int)(idx / HW2);        // b*2 + khi
    const int b = bk >> 1, khi = bk & 1;
    const float* xs = x + ((size_t)b * Nc + khi * 16) * HW2 + pix;
    _Float16* dst = xpack + idx * 16;
#pragma unroll
    for (int j = 0; j < 16; ++j) dst[j] = (_Float16)xs[(size_t)j * HW2];
}

// ---------------------------------------------------------------------------
// Kernel 1: attention1 via WMMA. Per wave: 16-pixel tile.
// att1[9 x 16pix] = A1[9(pad16) x 288] x patch[288 x 16], K split 9 x 32
// -> 9 v_wmma_f32_16x16x32_f16. Result transposed through LDS, then a tiny
// 9->9->9 MLP (ReLU / ReLU / sigmoid) per pixel on 16 lanes.
// ---------------------------------------------------------------------------
__global__ __launch_bounds__(256) void att_wmma_kernel(
    const _Float16* __restrict__ xpack,
    const _Float16* __restrict__ a1pack,
    const float* __restrict__ a1b,
    const float* __restrict__ a2w, const float* __restrict__ a2b,
    const float* __restrict__ a3w, const float* __restrict__ a3b,
    float* __restrict__ att)
{
    __shared__ float s_a2w[81], s_a3w[81];
    __shared__ float s_a1b[9], s_a2b[9], s_a3b[9];
    __shared__ float s_t[8][16][17];           // [wave][pixel][channel] padded
    if (threadIdx.x < 81) { s_a2w[threadIdx.x] = a2w[threadIdx.x];
                            s_a3w[threadIdx.x] = a3w[threadIdx.x]; }
    if (threadIdx.x < 9)  { s_a1b[threadIdx.x] = a1b[threadIdx.x];
                            s_a2b[threadIdx.x] = a2b[threadIdx.x];
                            s_a3b[threadIdx.x] = a3b[threadIdx.x]; }
    __syncthreads();

    const int lane = threadIdx.x & 31;
    const int wave = threadIdx.x >> 5;
    const int b       = blockIdx.x / 288;
    const int tileInB = (blockIdx.x % 288) * 8 + wave;   // 2304 tiles of 16 px
    const int pixBase = tileInB * 16;
    const int ph = pixBase / HW;
    const int w0 = pixBase % HW;

    const int col = lane & 15;
    const int khi = lane >> 4;
    const int pw  = w0 + col;

    const _Float16* xb = xpack + (size_t)(b * 2 + khi) * HW2 * 16;

    v8f acc = {};
#pragma unroll
    for (int p = 0; p < 9; ++p) {
        const int dh = p / 3 - 1, dw = p % 3 - 1;
        const int sh = ph + dh, sw = pw + dw;
        const int shc = clampi(sh, 0, HW - 1);
        const int swc = clampi(sw, 0, HW - 1);
        const _Float16 ok = ((sh >= 0) & (sh < HW) & (sw >= 0) & (sw < HW))
                                ? (_Float16)1.0f : (_Float16)0.0f;
        const v16h xh = *(const v16h*)(xb + (size_t)(shc * HW + swc) * 16);
        const v16h bf = xh * ok;            // zero-pad mask
        const v16h a0 = *(const v16h*)(a1pack + ((size_t)p * 32 + lane) * 16);
        acc = __builtin_amdgcn_wmma_f32_16x16x32_f16(
            false, a0, false, bf, (short)0, acc, false, false);
    }

    // Transpose D (channels spread over lanes/VGPRs) into LDS per pixel.
    const int rowBase = (lane < 16) ? 0 : 8;
#pragma unroll
    for (int r = 0; r < 8; ++r)
        s_t[wave][col][rowBase + r] = acc[r];
    __syncthreads();

    if (lane < 16) {
        float a1v[9], a2v[9];
#pragma unroll
        for (int o = 0; o < 9; ++o)
            a1v[o] = fmaxf(s_t[wave][col][o] + s_a1b[o], 0.0f);
#pragma unroll
        for (int o = 0; o < 9; ++o) {
            float t = s_a2b[o];
#pragma unroll
            for (int i = 0; i < 9; ++i) t = fmaf(s_a2w[o * 9 + i], a1v[i], t);
            a2v[o] = fmaxf(t, 0.0f);
        }
        float* ap = att + (size_t)b * 9 * HW2 + ph * HW + pw;
#pragma unroll
        for (int o = 0; o < 9; ++o) {
            float t = s_a3b[o];
#pragma unroll
            for (int i = 0; i < 9; ++i) t = fmaf(s_a3w[o * 9 + i], a2v[i], t);
            ap[(size_t)o * HW2] = 1.0f / (1.0f + __expf(-t));
        }
    }
}

// ---------------------------------------------------------------------------
// Kernel 2a: global average pool per (b, n). One block per (b,n), LDS reduce.
// ---------------------------------------------------------------------------
__global__ __launch_bounds__(256) void pool_kernel(const float* __restrict__ x,
                                                   float* __restrict__ pooled)
{
    __shared__ float red[256];
    const int bn = blockIdx.x;                  // b*32 + n
    const float* xp = x + (size_t)bn * HW2;
    float s = 0.0f;
    for (int i = threadIdx.x; i < HW2; i += 256) s += xp[i];
    red[threadIdx.x] = s;
    __syncthreads();
    for (int o = 128; o > 0; o >>= 1) {
        if (threadIdx.x < o) red[threadIdx.x] += red[threadIdx.x + o];
        __syncthreads();
    }
    if (threadIdx.x == 0) pooled[bn] = red[0] * (1.0f / (float)HW2);
}

// ---------------------------------------------------------------------------
// Kernel 2b: bias[b,m] = b2_b + b2_w @ relu(b1_w @ pooled + b1_b). 1 block.
// ---------------------------------------------------------------------------
__global__ __launch_bounds__(256) void bias_kernel(
    const float* __restrict__ pooled,
    const float* __restrict__ b1w, const float* __restrict__ b1b,
    const float* __restrict__ b2w, const float* __restrict__ b2b,
    float* __restrict__ bias)
{
    const int t = threadIdx.x;       // 256 = 8 batches * 32 channels
    const int b = t >> 5, m = t & 31;
    const float* pb = pooled + b * Nc;
    float acc = b2b[m];
    for (int j = 0; j < Mc; ++j) {
        float h = b1b[j];
#pragma unroll
        for (int n = 0; n < Nc; ++n) h = fmaf(pb[n], b1w[j * Nc + n], h);
        h = fmaxf(h, 0.0f);
        acc = fmaf(h, b2w[m * Mc + j], acc);
    }
    bias[t] = acc;
}

// ---------------------------------------------------------------------------
// Kernel 4: main einsum via WMMA. One wave = 32-pixel tile (two 16-col B
// tiles sharing the same A fragments). D[32x16] = W[32x288] x B[288x16],
// K split 9 x 32 -> 36 v_wmma_f32_16x16x32_f16 per wave. B fragments come
// from channel-interleaved f16 (2 x b128 per fragment) scaled by attention
// via packed f16 muls; the scale carries the zero-padding mask.
// ---------------------------------------------------------------------------
__global__ __launch_bounds__(256) void laconv_main(
    const _Float16* __restrict__ xpack,
    const _Float16* __restrict__ wpack,
    const float* __restrict__ att,
    const float* __restrict__ bias,
    float* __restrict__ out)
{
    const int lane = threadIdx.x & 31;
    const int wave = threadIdx.x >> 5;
    const int b       = blockIdx.x / 144;
    const int tileInB = (blockIdx.x % 144) * 8 + wave;   // 1152 tiles of 32 px
    const int pixBase = tileInB * 32;
    const int ph = pixBase / HW;
    const int w0 = pixBase % HW;          // tiles never cross a row (32 | 192)

    // Stage the per-batch bias vector into LDS via the CDNA5 async-LDS path.
    __shared__ float s_bias[Mc];
    if (threadIdx.x < Mc) {
        const unsigned ldsAddr =
            (unsigned)(unsigned long long)(&s_bias[threadIdx.x]);
        const float* g = bias + b * Mc + threadIdx.x;
        asm volatile("global_load_async_to_lds_b32 %0, %1, off"
                     :: "v"(ldsAddr), "v"(g) : "memory");
    }
    asm volatile("s_wait_asynccnt 0x0" ::: "memory");
    __syncthreads();

    const int col = lane & 15;            // B-matrix column = pixel in tile
    const int khi = lane >> 4;            // channel half (0..15 / 16..31)
    const int pw0 = w0 + col;
    const int pw1 = w0 + 16 + col;

    // attention values for both pixel columns, 9 kernel positions
    float attv0[9], attv1[9];
    const float* attBase = att + (size_t)b * 9 * HW2 + ph * HW;
#pragma unroll
    for (int p = 0; p < 9; ++p) {
        attv0[p] = attBase[(size_t)p * HW2 + pw0];
        attv1[p] = attBase[(size_t)p * HW2 + pw1];
    }

    const _Float16* xb = xpack + (size_t)(b * 2 + khi) * HW2 * 16;
    __builtin_prefetch(xb + (size_t)(ph * HW + pw0) * 16, 0, 3);

    v8f acc00 = {}, acc01 = {};   // pixel tile 0, m 0-15 / 16-31
    v8f acc10 = {}, acc11 = {};   // pixel tile 1

#pragma unroll
    for (int p = 0; p < 9; ++p) {
        const int dh = p / 3 - 1, dw = p % 3 - 1;
        const int sh    = ph + dh;
        const int shc   = clampi(sh, 0, HW - 1);
        const bool rowOk = (sh >= 0) & (sh < HW);

        const int sw0   = pw0 + dw;
        const int sw0c  = clampi(sw0, 0, HW - 1);
        const float sc0 = (rowOk & (sw0 >= 0) & (sw0 < HW)) ? attv0[p] : 0.0f;

        const int sw1   = pw1 + dw;
        const int sw1c  = clampi(sw1, 0, HW - 1);
        const float sc1 = (rowOk & (sw1 >= 0) & (sw1 < HW)) ? attv1[p] : 0.0f;

        // 16-channel f16 vectors for the (clamped) source pixels: 2 x b128.
        const v16h xh0 = *(const v16h*)(xb + (size_t)(shc * HW + sw0c) * 16);
        const v16h xh1 = *(const v16h*)(xb + (size_t)(shc * HW + sw1c) * 16);

        // B fragments: packed f16 scale by attention (mask folded into scale).
        const v16h bf0 = xh0 * (_Float16)sc0;
        const v16h bf1 = xh1 * (_Float16)sc1;

        const v16h a0 = *(const v16h*)(wpack + ((size_t)(p * 2 + 0) * 32 + lane) * 16);
        const v16h a1 = *(const v16h*)(wpack + ((size_t)(p * 2 + 1) * 32 + lane) * 16);

        acc00 = __builtin_amdgcn_wmma_f32_16x16x32_f16(
            false, a0, false, bf0, (short)0, acc00, false, false);
        acc01 = __builtin_amdgcn_wmma_f32_16x16x32_f16(
            false, a1, false, bf0, (short)0, acc01, false, false);
        acc10 = __builtin_amdgcn_wmma_f32_16x16x32_f16(
            false, a0, false, bf1, (short)0, acc10, false, false);
        acc11 = __builtin_amdgcn_wmma_f32_16x16x32_f16(
            false, a1, false, bf1, (short)0, acc11, false, false);
    }

    // Epilogue: 16x16 f32 C/D layout -> lane<16: M=r, lane>=16: M=r+8; N=lane&15
    const int mrowBase = (lane < 16) ? 0 : 8;
    float* ob0 = out + (size_t)b * Mc * HW2 + ph * HW + pw0;
    float* ob1 = ob0 + 16;
#pragma unroll
    for (int r = 0; r < 8; ++r) {
        const int m0 = mrowBase + r;
        const int m1 = 16 + mrowBase + r;
        const float bm0 = s_bias[m0];
        const float bm1 = s_bias[m1];
        ob0[(size_t)m0 * HW2] = acc00[r] + bm0;
        ob0[(size_t)m1 * HW2] = acc01[r] + bm1;
        ob1[(size_t)m0 * HW2] = acc10[r] + bm0;
        ob1[(size_t)m1 * HW2] = acc11[r] + bm1;
    }
}

// ---------------------------------------------------------------------------
// Launch
// ---------------------------------------------------------------------------
extern "C" void kernel_launch(void* const* d_in, const int* in_sizes, int n_in,
                              void* d_out, int out_size, void* d_ws, size_t ws_size,
                              hipStream_t stream)
{
    const float* x    = (const float*)d_in[0];
    const float* a1w  = (const float*)d_in[1];
    const float* a1b  = (const float*)d_in[2];
    const float* a2w  = (const float*)d_in[3];
    const float* a2b  = (const float*)d_in[4];
    const float* a3w  = (const float*)d_in[5];
    const float* a3b  = (const float*)d_in[6];
    const float* b1w  = (const float*)d_in[7];
    const float* b1b  = (const float*)d_in[8];
    const float* b2w  = (const float*)d_in[9];
    const float* b2b  = (const float*)d_in[10];
    const float* wght = (const float*)d_in[11];
    float* out = (float*)d_out;

    // Workspace layout (floats from base):
    //   att    [8*9*HW2]            = 2,654,208 floats
    //   pooled [256], bias [256]
    //   wpack  (f16) 9*2*32*16      = 9,216 halves (4,608 floats)
    //   a1pack (f16) 9*32*16        = 4,608 halves (2,304 floats)
    //   xpack  (f16) 8*2*HW2*16     = 9,437,184 halves (32B-aligned)
    float* ws       = (float*)d_ws;
    float* att      = ws;
    float* pooled   = ws + (size_t)Bn * 9 * HW2;
    float* bias     = pooled + 256;
    _Float16* wpck  = (_Float16*)(ws + (size_t)Bn * 9 * HW2 + 512);
    _Float16* a1pck = (_Float16*)(ws + (size_t)Bn * 9 * HW2 + 512 + 4608);
    _Float16* xpck  = (_Float16*)(ws + (size_t)Bn * 9 * HW2 + 512 + 4608 + 2304);

    pack_kernel<<<1, 32, 0, stream>>>(wght, a1w, wpck, a1pck);
    xpack_kernel<<<Bn * 2 * HW2 / 256, 256, 0, stream>>>(x, xpck);
    att_wmma_kernel<<<Bn * 288, 256, 0, stream>>>(xpck, a1pck, a1b,
                                                  a2w, a2b, a3w, a3b, att);
    pool_kernel<<<Bn * Nc, 256, 0, stream>>>(x, pooled);
    bias_kernel<<<1, 256, 0, stream>>>(pooled, b1w, b1b, b2w, b2b, bias);
    laconv_main<<<Bn * 144, 256, 0, stream>>>(xpck, wpck, att, bias, out);
}
